// GraphAttentionLayer_25331717112048
// MI455X (gfx1250) — matrix operand order repaired and checked by reading, not compiled, not run on hardware
//
#include <hip/hip_runtime.h>
#include <hip/hip_bf16.h>

typedef __attribute__((ext_vector_type(16))) _Float16 v16h;
typedef __attribute__((ext_vector_type(8)))  _Float16 v8h;
typedef __attribute__((ext_vector_type(8)))  float    v8f;
typedef __attribute__((ext_vector_type(4)))  float    v4f;

#define B_BATCH 32768
#define N_NODES 15
#define FIN 128
#define FOUT 128
#define ROWS (B_BATCH * N_NODES)   // 491520 rows = 30720 row-tiles of 16
#define WT_STRIDE 136              // halfs per W^T row in LDS (bank-spread pad)
#define GEMM_BLOCKS 960            // 960 blocks * 4 stripe-slots * 8 iters = 30720
#define TILE_STRIDE (GEMM_BLOCKS * 4)

// ---------------------------------------------------------------------------
// Kernel 1: H(f16) = f16(X) * f16(W), f32 accumulate via v_wmma_f32_16x16x32_f16.
// 256 threads = 8 waves. Wave pair splits the 128 output columns: each wave
// owns 4 N-tiles and keeps ALL its B fragments (4 nt x 4 kc = 128 VGPRs)
// resident after a single LDS read. Steady state: global A loads + cvt +
// 16 WMMA + f16 stores; next tile's contiguous 8KB is prefetched each iter.
// ---------------------------------------------------------------------------
__global__ __launch_bounds__(256)
void gat_gemm_wmma(const float* __restrict__ X, const float* __restrict__ W,
                   _Float16* __restrict__ H)
{
    __shared__ _Float16 WT[128 * WT_STRIDE];   // W^T as f16: WT[n][k]

    const int t = threadIdx.x;
    for (int it = 0; it < 64; ++it) {          // stage W (row-major) transposed
        int idx = t + 256 * it;                // 16384 elements, coalesced reads
        int k = idx >> 7, n = idx & 127;
        WT[n * WT_STRIDE + k] = (_Float16)W[idx];
    }
    __syncthreads();

    const int lane = t & 31;
    const int wave = t >> 5;
    const int n    = lane & 15;
    const int hi   = lane >> 4;
    const int ch   = wave & 1;                 // column half: nt in [4*ch, 4*ch+4)
    const int wsl  = wave >> 1;                // row-stripe slot 0..3

    // --- B fragments: registers for the whole kernel lifetime --------------
    v16h bfrag[4][4];                          // [nt4][kc]
    #pragma unroll
    for (int nt4 = 0; nt4 < 4; ++nt4)
        #pragma unroll
        for (int kc = 0; kc < 4; ++kc) {
            const _Float16* bp =
                &WT[((ch * 4 + nt4) * 16 + n) * WT_STRIDE + kc * 32 + 8 * hi];
            v8h b0 = *(const v8h*)(bp);
            v8h b1 = *(const v8h*)(bp + 16);
            v16h b;
            #pragma unroll
            for (int e = 0; e < 8; ++e) { b[e] = b0[e]; b[e + 8] = b1[e]; }
            bfrag[nt4][kc] = b;
        }

    // --- 8 M-tiles per wave, exact cover, no divergence ---------------------
    for (int it = 0; it < 8; ++it) {
        const long tile = (long)blockIdx.x * 4 + wsl + (long)it * TILE_STRIDE;
        const long m0   = tile * 16;
        const float* xrow = X + (m0 + n) * FIN;

        // Prefetch next tile: its 16 rows are 8KB contiguous; 32 lanes x 256B.
        if (it < 7) {
            const char* nx = (const char*)(X + (m0 + (long)TILE_STRIDE * 16) * FIN);
            __builtin_prefetch(nx + lane * 256, 0, 3);
        }

        v16h afrag[4];
        #pragma unroll
        for (int kc = 0; kc < 4; ++kc) {       // A: lane holds K = kc*32+8hi+0..7, +16
            const float* p0 = xrow + kc * 32 + 8 * hi;
            v4f f0 = *(const v4f*)(p0);
            v4f f1 = *(const v4f*)(p0 + 4);
            v4f f2 = *(const v4f*)(p0 + 16);
            v4f f3 = *(const v4f*)(p0 + 20);
            v16h a;
            #pragma unroll
            for (int e = 0; e < 4; ++e) {
                a[e]      = (_Float16)f0[e];
                a[e + 4]  = (_Float16)f1[e];
                a[e + 8]  = (_Float16)f2[e];
                a[e + 12] = (_Float16)f3[e];
            }
            afrag[kc] = a;
        }

        v8f acc[4];
        #pragma unroll
        for (int nt4 = 0; nt4 < 4; ++nt4)
            #pragma unroll
            for (int r = 0; r < 8; ++r) acc[nt4][r] = 0.0f;

        #pragma unroll
        for (int kc = 0; kc < 4; ++kc)         // 4 independent acc chains per kc
            #pragma unroll
            for (int nt4 = 0; nt4 < 4; ++nt4)
                acc[nt4] = __builtin_amdgcn_wmma_f32_16x16x32_f16(
                    false, afrag[kc], false, bfrag[nt4][kc],
                    (short)0, acc[nt4], false, false);

        #pragma unroll
        for (int nt4 = 0; nt4 < 4; ++nt4)
            #pragma unroll
            for (int r = 0; r < 8; ++r)
                H[(m0 + r + 8 * hi) * FOUT + ch * 64 + nt4 * 16 + n] =
                    (_Float16)acc[nt4][r];
    }
}

// ---------------------------------------------------------------------------
// Kernel 2: per-batch attention. Each thread owns feature column f and keeps
// h[b,:,f] in registers; s_src/s_dst come from a shfl+LDS reduction of those
// same registers. Masked leaky-relu scores, softmax over axis i (per column
// j, faithful to reference), h' = elu(att @ h).
// ---------------------------------------------------------------------------
__global__ __launch_bounds__(128)
void gat_attention(const _Float16* __restrict__ H, const int* __restrict__ adj,
                   const float* __restrict__ a_src, const float* __restrict__ a_dst,
                   float* __restrict__ out)
{
    __shared__ float att[N_NODES][16];
    __shared__ float ss[N_NODES], sd[N_NODES];
    __shared__ float red_s[4][N_NODES], red_d[4][N_NODES];

    const long b = blockIdx.x;
    const int  t = threadIdx.x;
    const int  f = t;
    const int  wave = t >> 5, lane = t & 31;

    float hv[N_NODES];
    #pragma unroll
    for (int j = 0; j < N_NODES; ++j)
        hv[j] = (float)H[(b * N_NODES + j) * FOUT + f];

    const float pa = a_src[f], pb = a_dst[f];
    #pragma unroll
    for (int j = 0; j < N_NODES; ++j) {        // block-reduce hv[j]*a over f
        float s = hv[j] * pa, d = hv[j] * pb;
        #pragma unroll
        for (int m = 1; m < 32; m <<= 1) {
            s += __shfl_xor(s, m, 32);
            d += __shfl_xor(d, m, 32);
        }
        if (lane == 0) { red_s[wave][j] = s; red_d[wave][j] = d; }
    }
    __syncthreads();
    if (t < N_NODES) {
        float s = 0.f, d = 0.f;
        #pragma unroll
        for (int w = 0; w < 4; ++w) { s += red_s[w][t]; d += red_d[w][t]; }
        ss[t] = s; sd[t] = d;
    }
    __syncthreads();

    for (int u = t; u < N_NODES * N_NODES; u += 128) {
        int i = u / N_NODES, j = u - i * N_NODES;
        float e  = ss[i] + sd[j];
        float lr = e > 0.f ? e : 0.2f * e;               // leaky_relu(0.2)
        att[i][j] = (adj[b * (N_NODES * N_NODES) + u] > 0) ? lr : -9e15f;
    }
    __syncthreads();

    if (t < N_NODES) {                                   // softmax over i, column j=t
        const int j = t;
        float m = -3.4e38f;
        #pragma unroll
        for (int i = 0; i < N_NODES; ++i) m = fmaxf(m, att[i][j]);
        float ex[N_NODES], s = 0.f;
        #pragma unroll
        for (int i = 0; i < N_NODES; ++i) { ex[i] = __expf(att[i][j] - m); s += ex[i]; }
        float inv = 1.f / s;
        #pragma unroll
        for (int i = 0; i < N_NODES; ++i) att[i][j] = ex[i] * inv;
    }
    __syncthreads();

    #pragma unroll
    for (int i = 0; i < N_NODES; ++i) {
        float acc = 0.f;
        #pragma unroll
        for (int j = 0; j < N_NODES; ++j) acc = fmaf(att[i][j], hv[j], acc);
        out[(b * N_NODES + i) * FOUT + f] = acc > 0.f ? acc : expm1f(acc);  // elu
    }
}

// ---------------------------------------------------------------------------
extern "C" void kernel_launch(void* const* d_in, const int* in_sizes, int n_in,
                              void* d_out, int out_size, void* d_ws, size_t ws_size,
                              hipStream_t stream) {
    const float* x     = (const float*)d_in[0];   // [B,N,Fin]
    const int*   adj   = (const int*)  d_in[1];   // [B,N,N]
    const float* W     = (const float*)d_in[2];   // [Fin,Fout]
    const float* a_src = (const float*)d_in[3];   // [Fout]
    const float* a_dst = (const float*)d_in[4];   // [Fout]
    float* out = (float*)d_out;

    _Float16* H = (_Float16*)d_ws;                // ROWS*FOUT f16 scratch (126MB)

    gat_gemm_wmma<<<GEMM_BLOCKS, 256, 0, stream>>>(x, W, H);
    gat_attention<<<B_BATCH, 128, 0, stream>>>(H, adj, a_src, a_dst, out);
}
